// GNN_skip_small_10273561772114
// MI455X (gfx1250) — compile-verified, matrix operand order checked by reading, not run on hardware
//
#include <hip/hip_runtime.h>

// ---------- CDNA5 WMMA types ----------
typedef __attribute__((ext_vector_type(16))) __bf16 v16bf;
typedef __attribute__((ext_vector_type(8)))  float  v8f;

#define GNN_N 8192
#define GNN_F 128
#define GNN_G 256
#define GNN_NF (GNN_N * GNN_F)
#define TILE_M 128
#define TILE_K 32
#define KSPLIT 8

// round-half-up fp32 -> bf16 (bias the bit pattern, keep high 16)
static __device__ __forceinline__ unsigned short f2bf_hi(float f) {
  return (unsigned short)((__float_as_uint(f) + 0x8000u) >> 16);
}
// pack two fp32 -> packed bf16 dword {hi,lo} : 2x v_add + 1x v_perm_b32
static __device__ __forceinline__ unsigned pk2bf(float lo, float hi) {
  unsigned ulo = __float_as_uint(lo) + 0x8000u;
  unsigned uhi = __float_as_uint(hi) + 0x8000u;
  return __builtin_amdgcn_perm(uhi, ulo, 0x07060302u); // bytes: uhi[3:2] | ulo[3:2]
}
// Low 32 bits of a generic LDS pointer == LDS byte address.
static __device__ __forceinline__ unsigned lds_addr(const void* p) {
  return (unsigned)(unsigned long long)p;
}
static __device__ __forceinline__ void async_ld_b128_nt(unsigned lds, const float* g) {
  // adj stream is single-use (256 MB > 192 MB L2): non-temporal keeps L2 for B/partials
  asm volatile("global_load_async_to_lds_b128 %0, %1, off th:TH_LOAD_NT"
               :: "v"(lds), "v"(g) : "memory");
}
static __device__ __forceinline__ void wait_asynccnt0() {
  asm volatile("s_wait_asynccnt 0x0" ::: "memory");
}

union FragBF { v16bf v; unsigned int u[8]; };

// C[128 x 128] tile = A_cat[128 x K] * B[K x 128], bf16 WMMA, f32 accum.
// A cols [0,K0) from A0 (stride lda0), [K0,Ktot) from A1 (stride lda1).
// blockIdx.x = row tile, blockIdx.y = K split (writes to C + y*cSplitStride).
// A tiles: fp32 via GLOBAL_LOAD_ASYNC_TO_LDS_B128 (double-buffered, ASYNCcnt),
//          converted to bf16 at fragment-build time (v_perm pack).
// B tiles: fp32 global loads -> registers -> transposed bf16 LDS (double-buffered).
// Optional fused epilogue: relu(c + bias) * bn_scale + bn_shift (BN after ReLU).
__global__ __launch_bounds__(256)
void gnn_gemm128_bf16(const float* __restrict__ A0, long lda0, int K0,
                      const float* __restrict__ A1, long lda1,
                      const float* __restrict__ B,
                      float* __restrict__ C, long cSplitStride,
                      int Ktot, int kChunk,
                      const float* __restrict__ bias,
                      const float* __restrict__ bng, const float* __restrict__ bnb,
                      const float* __restrict__ bnm, const float* __restrict__ bnv,
                      int do_epi)
{
  __shared__ __align__(16) float        sA[2][TILE_M * TILE_K];      // 2 x 16 KB fp32
  __shared__ __align__(16) unsigned int sB[2][128 * TILE_K / 2];     // 2 x 8 KB  bf16 (B^T)

  const int tid  = threadIdx.x;
  const int lane = tid & 31;       // wave32
  const int wave = tid >> 5;       // 8 waves; wave = 16 rows x 128 cols
  const int half = lane >> 4;
  const int mlan = lane & 15;
  const long mBase = (long)blockIdx.x * TILE_M;

  const int kBegin = blockIdx.y * kChunk;
  int kEnd = kBegin + kChunk; if (kEnd > Ktot) kEnd = Ktot;
  C += (long)blockIdx.y * cSplitStride;

  // per-thread tile-copy coordinates (128x32 tile, 4 float4 per thread)
  const int aRow0 = tid >> 3;                  // +32 per j
  const int aC4   = tid & 7;
  const int bKr0  = tid >> 5;                  // +8 per j
  const int bN0   = (tid & 31) * 4;

  v8f acc[8] = {};
  float4 breg[4];

  // ---- issue async A copies for stage stg at k ----
  auto issueA = [&](int k, int stg) {
    int kk = k + aC4 * 4;
    #pragma unroll
    for (int j = 0; j < 4; ++j) {
      int row = aRow0 + j * 32;
      const float* src = (kk < K0) ? (A0 + (mBase + row) * lda0 + kk)
                                   : (A1 + (mBase + row) * lda1 + (kk - K0));
      async_ld_b128_nt(lds_addr(&sA[stg][row * TILE_K + aC4 * 4]), src);
    }
  };
  // ---- start B tile loads into registers ----
  auto loadB = [&](int k) {
    #pragma unroll
    for (int j = 0; j < 4; ++j)
      breg[j] = *(const float4*)(B + (long)(k + bKr0 + j * 8) * 128 + bN0);
  };
  // ---- convert + transpose-store B registers into bf16 LDS stage ----
  auto storeB = [&](int stg) {
    unsigned short* bt = (unsigned short*)sB[stg];
    #pragma unroll
    for (int j = 0; j < 4; ++j) {
      int kr = bKr0 + j * 8;
      bt[(bN0 + 0) * TILE_K + kr] = f2bf_hi(breg[j].x);  // ds_store_b16_d16_hi
      bt[(bN0 + 1) * TILE_K + kr] = f2bf_hi(breg[j].y);
      bt[(bN0 + 2) * TILE_K + kr] = f2bf_hi(breg[j].z);
      bt[(bN0 + 3) * TILE_K + kr] = f2bf_hi(breg[j].w);
    }
  };

  // ---- prologue: fill stage 0 ----
  issueA(kBegin, 0);
  loadB(kBegin);
  storeB(0);
  wait_asynccnt0();
  __syncthreads();

  int s = 0;
  for (int k = kBegin; k < kEnd; k += TILE_K) {
    const int nk = k + TILE_K;
    const bool more = nk < kEnd;
    if (more) { issueA(nk, s ^ 1); loadB(nk); }  // overlap with compute below

    // ---- A fragment: read fp32 stage, pack to bf16 (documented 16x32 layout:
    // lane -> row M = lane%16; u[0..3]: K = half*8 + {0..7}; u[4..7]: K = 16 + half*8 + {0..7})
    const float* aRowP = &sA[s][(wave * 16 + mlan) * TILE_K];
    float4 a0 = *(const float4*)(aRowP + half * 8);
    float4 a1 = *(const float4*)(aRowP + half * 8 + 4);
    float4 a2 = *(const float4*)(aRowP + 16 + half * 8);
    float4 a3 = *(const float4*)(aRowP + 16 + half * 8 + 4);
    FragBF fa;
    fa.u[0] = pk2bf(a0.x, a0.y); fa.u[1] = pk2bf(a0.z, a0.w);
    fa.u[2] = pk2bf(a1.x, a1.y); fa.u[3] = pk2bf(a1.z, a1.w);
    fa.u[4] = pk2bf(a2.x, a2.y); fa.u[5] = pk2bf(a2.z, a2.w);
    fa.u[6] = pk2bf(a3.x, a3.y); fa.u[7] = pk2bf(a3.z, a3.w);

    const int fb = mlan * 16 + half * 4;  // dword offset within a 16-row bf16 tile
    #pragma unroll
    for (int nt = 0; nt < 8; ++nt) {
      FragBF fbm;
      const unsigned int* bBase = sB[s] + nt * 256;  // cols nt*16..+15 (B^T rows)
      #pragma unroll
      for (int i = 0; i < 4; ++i) { fbm.u[i] = bBase[fb + i]; fbm.u[i + 4] = bBase[fb + 8 + i]; }
      acc[nt] = __builtin_amdgcn_wmma_f32_16x16x32_bf16(
          false, fa.v, false, fbm.v, (short)0, acc[nt], false, false);
    }

    if (more) storeB(s ^ 1);
    wait_asynccnt0();     // my stage s^1 async copies have landed
    __syncthreads();      // everyone's stage s^1 A + B visible
    s ^= 1;
  }

  // ---- epilogue + store: C/D layout: VGPR r, lane -> row = r + half*8, col = lane%16 ----
  #pragma unroll
  for (int nt = 0; nt < 8; ++nt) {
    int n = nt * 16 + mlan;
    float sc = 1.f, sh = 0.f, bi = 0.f;
    if (do_epi) {
      bi = bias[n];
      float t = bng[n] * rsqrtf(bnv[n] + 1e-5f);
      sc = t;
      sh = bnb[n] - bnm[n] * t;
    }
    #pragma unroll
    for (int r = 0; r < 8; ++r) {
      long row = mBase + wave * 16 + half * 8 + r;
      float c = acc[nt][r];
      if (do_epi) c = fmaxf(c + bi, 0.f) * sc + sh;
      C[row * 128 + n] = c;
    }
  }
}

// Deterministic reduction of KSPLIT partial C buffers.
__global__ __launch_bounds__(256)
void gnn_reduce_parts(const float* __restrict__ part, float* __restrict__ out, int n)
{
  int i = blockIdx.x * 256 + threadIdx.x;
  if (i < n) {
    float s = 0.f;
    #pragma unroll
    for (int j = 0; j < KSPLIT; ++j) s += part[(long)j * GNN_NF + i];
    out[i] = s;
  }
}

// Per-node scalar score s[n] = [x_in|x1|x2][n] . W4, segment-summed into acc[G].
__global__ __launch_bounds__(256)
void gnn_node_score(const float* __restrict__ x0, const float* __restrict__ x1,
                    const float* __restrict__ x2, const float* __restrict__ W4,
                    const int* __restrict__ idx, float* __restrict__ acc)
{
  int node = blockIdx.x * 8 + (threadIdx.x >> 5);
  int lane = threadIdx.x & 31;
  if (node >= GNN_N) return;
  float s = 0.f;
  #pragma unroll
  for (int j0 = 0; j0 < GNN_F; j0 += 32) {
    int j = j0 + lane;
    s += x0[(long)node * GNN_F + j] * W4[j];
    s += x1[(long)node * GNN_F + j] * W4[GNN_F + j];
    s += x2[(long)node * GNN_F + j] * W4[2 * GNN_F + j];
  }
  #pragma unroll
  for (int off = 16; off > 0; off >>= 1) s += __shfl_down(s, off, 32);
  if (lane == 0) atomicAdd(&acc[idx[node]], s);
}

__global__ void gnn_zero(float* __restrict__ p, int n)
{
  int i = blockIdx.x * 256 + threadIdx.x;
  if (i < n) p[i] = 0.f;
}

__global__ void gnn_finalize(const float* __restrict__ acc, const float* __restrict__ b4,
                             float* __restrict__ out)
{
  int g = threadIdx.x;
  if (g < GNN_G) out[g] = fmaxf(acc[g] + b4[0], 0.f);
}

extern "C" void kernel_launch(void* const* d_in, const int* in_sizes, int n_in,
                              void* d_out, int out_size, void* d_ws, size_t ws_size,
                              hipStream_t stream) {
  const float* adj  = (const float*)d_in[0];
  const float* x_in = (const float*)d_in[1];
  const float* W1   = (const float*)d_in[2];
  const float* b1   = (const float*)d_in[3];
  const float* W2   = (const float*)d_in[4];
  const float* b2   = (const float*)d_in[5];
  const float* W4   = (const float*)d_in[6];
  const float* b4   = (const float*)d_in[7];
  const float* bn1g = (const float*)d_in[8];
  const float* bn1b = (const float*)d_in[9];
  const float* bn1m = (const float*)d_in[10];
  const float* bn1v = (const float*)d_in[11];
  const float* bn2g = (const float*)d_in[12];
  const float* bn2b = (const float*)d_in[13];
  const float* bn2m = (const float*)d_in[14];
  const float* bn2v = (const float*)d_in[15];
  const int*   idx  = (const int*)d_in[16];
  (void)in_sizes; (void)n_in; (void)out_size;

  float* ws  = (float*)d_ws;
  float* h0  = ws;                          // [N,F]
  float* x1  = ws + (size_t)GNN_NF;         // [N,H1]
  float* h1  = ws + 2 * (size_t)GNN_NF;     // [N,H1]
  float* x2  = ws + 3 * (size_t)GNN_NF;     // [N,H2]
  float* acc = ws + 4 * (size_t)GNN_NF;     // [G]
  float* prt = acc + GNN_G;                 // [KSPLIT,N,F] split-K partials
  if (ws_size < ((4 + KSPLIT) * (size_t)GNN_NF + GNN_G) * sizeof(float)) return;

  const dim3 tB(256);
  const dim3 gSplit(GNN_N / TILE_M, KSPLIT);  // 512 WGs per adj pass
  const dim3 gOne(GNN_N / TILE_M, 1);
  const int kChunkBig = GNN_N / KSPLIT;       // 1024

  gnn_zero<<<1, 256, 0, stream>>>(acc, GNN_G);

  // h0 = adj @ x_in   (split-K partials, deterministic reduce)
  gnn_gemm128_bf16<<<gSplit, tB, 0, stream>>>(adj, GNN_N, GNN_N, nullptr, 0,
      x_in, prt, GNN_NF, GNN_N, kChunkBig,
      nullptr, nullptr, nullptr, nullptr, nullptr, 0);
  gnn_reduce_parts<<<GNN_NF / 256, 256, 0, stream>>>(prt, h0, GNN_NF);

  // x1 = BN1(relu([x_in | h0] @ W1 + b1))
  gnn_gemm128_bf16<<<gOne, tB, 0, stream>>>(x_in, GNN_F, GNN_F, h0, GNN_F,
      W1, x1, 0, 2 * GNN_F, 2 * GNN_F,
      b1, bn1g, bn1b, bn1m, bn1v, 1);

  // h1 = adj @ x1
  gnn_gemm128_bf16<<<gSplit, tB, 0, stream>>>(adj, GNN_N, GNN_N, nullptr, 0,
      x1, prt, GNN_NF, GNN_N, kChunkBig,
      nullptr, nullptr, nullptr, nullptr, nullptr, 0);
  gnn_reduce_parts<<<GNN_NF / 256, 256, 0, stream>>>(prt, h1, GNN_NF);

  // x2 = BN2(relu([x1 | h1] @ W2 + b2))
  gnn_gemm128_bf16<<<gOne, tB, 0, stream>>>(x1, GNN_F, GNN_F, h1, GNN_F,
      W2, x2, 0, 2 * GNN_F, 2 * GNN_F,
      b2, bn2g, bn2b, bn2m, bn2v, 1);

  // readout: segment_sum(xc @ W4) == pooled @ W4 (both linear), then relu(+b4)
  gnn_node_score<<<GNN_N / 8, 256, 0, stream>>>(x_in, x1, x2, W4, idx, acc);
  gnn_finalize<<<1, 256, 0, stream>>>(acc, b4, (float*)d_out);
}
